// GNNModel_57389353009708
// MI455X (gfx1250) — matrix operand order repaired and checked by reading, not compiled, not run on hardware
//
#include <hip/hip_runtime.h>
#include <hip/hip_bf16.h>
#include <math.h>

#define N_NODES 50000
#define N_EDGES 800000
#define N_GRAPHS 50

typedef __attribute__((ext_vector_type(16))) __bf16 bf16x16;
typedef __attribute__((ext_vector_type(8)))  float  f32x8;

union BFrag { bf16x16 v; __bf16 h[16]; uint4 q[2]; };

// native f32 -> bf16 convert (RNE); should lower to v_cvt_*bf16_f32 on gfx1250
__device__ __forceinline__ __bf16 f2bf(float f) { return (__bf16)f; }

// order-preserving float <-> uint for atomic max
__device__ __forceinline__ unsigned f2ord(float f) {
  unsigned u = __float_as_uint(f);
  return (u & 0x80000000u) ? ~u : (u | 0x80000000u);
}
__device__ __forceinline__ float ord2f(unsigned u) {
  return (u & 0x80000000u) ? __uint_as_float(u & 0x7FFFFFFFu) : __uint_as_float(~u);
}

// ---------------------------------------------------------------------------
// Y[n x M] = act(X[n x K] @ W[K x M] + bias) [+ pe tiled by 32 channels]
// 4 waves / block; each wave owns one 16x16 tile (v_wmma_f32_16x16x32_bf16).
// B tile staged once per block in LDS as bf16 (padded, conflict-free).
// K, M compile-time: K%32==0 path has zero bounds checks -> b128 loads.
// ---------------------------------------------------------------------------
template <int K, int M, bool RELU, bool ADDPE>
__global__ void gemm_bf16_wmma(const float* __restrict__ X,
                               const float* __restrict__ W,
                               const float* __restrict__ bias,
                               float* __restrict__ Y,
                               int n, const float* __restrict__ pe)
{
  constexpr int KP = K + 8;                 // bf16 row stride (16B-aligned rows)
  __shared__ __bf16 sW[16 * KP];

  const int lane    = threadIdx.x & 31;
  const int wave    = threadIdx.x >> 5;
  const int col_lo  = lane & 15;
  const int half    = lane >> 4;            // 0 or 1
  const int colbase = blockIdx.y * 16;

  // ---- stage W[:, colbase:colbase+16] as bf16 into LDS, layout sW[nn*KP + k]
  for (int i = threadIdx.x; i < K * 16; i += 128) {
    int k  = i >> 4;
    int nn = i & 15;                        // consecutive threads -> consecutive cols
    sW[nn * KP + k] = f2bf(W[(long)k * M + colbase + nn]);
  }
  __syncthreads();

  const int rowbase = (blockIdx.x * 4 + wave) * 16;
  if (rowbase >= n) return;                 // wave-uniform exit (after barrier)

  const int  r  = rowbase + col_lo;
  const int  rc = (r < n) ? r : (n - 1);
  const float* __restrict__ Xrow = X + (long)rc * K;

  const int kbaseA = half * 8;              // A: lanes 0-15 K{0..7,16..23}; +8 for hi
  const int kbaseB = half * 8;              // B: lanes 0-15 K{0..7,16..23}; +8 for hi

  f32x8 acc = {};

#pragma unroll
  for (int k0 = 0; k0 < K; k0 += 32) {
    BFrag a, b;
    if constexpr (K % 32 == 0) {
      // A fragment: 4 unconditional float4 loads (global_load_b128)
      float4 q0 = *(const float4*)(Xrow + k0 + kbaseA);
      float4 q1 = *(const float4*)(Xrow + k0 + kbaseA + 4);
      float4 q2 = *(const float4*)(Xrow + k0 + kbaseA + 16);
      float4 q3 = *(const float4*)(Xrow + k0 + kbaseA + 20);
      a.h[0]  = f2bf(q0.x); a.h[1]  = f2bf(q0.y); a.h[2]  = f2bf(q0.z); a.h[3]  = f2bf(q0.w);
      a.h[4]  = f2bf(q1.x); a.h[5]  = f2bf(q1.y); a.h[6]  = f2bf(q1.z); a.h[7]  = f2bf(q1.w);
      a.h[8]  = f2bf(q2.x); a.h[9]  = f2bf(q2.y); a.h[10] = f2bf(q2.z); a.h[11] = f2bf(q2.w);
      a.h[12] = f2bf(q3.x); a.h[13] = f2bf(q3.y); a.h[14] = f2bf(q3.z); a.h[15] = f2bf(q3.w);
      // B fragment: two 8-bf16 runs at +0 and +16 from LDS (2 x ds_load_b128)
      const __bf16* ps = &sW[col_lo * KP + k0 + kbaseB];
      b.q[0] = *(const uint4*)(ps);
      b.q[1] = *(const uint4*)(ps + 16);
    } else {
      // only the tiny K=16 layer takes this path
#pragma unroll
      for (int j = 0; j < 8; ++j) {
        int koff = kbaseA + ((j < 4) ? (2 * j) : (16 + 2 * (j - 4)));
        int k = k0 + koff;
        a.h[2 * j]     = (k     < K) ? f2bf(Xrow[k])     : (__bf16)0.f;
        a.h[2 * j + 1] = (k + 1 < K) ? f2bf(Xrow[k + 1]) : (__bf16)0.f;
      }
#pragma unroll
      for (int e = 0; e < 16; ++e) {
        int k = k0 + kbaseB + ((e < 8) ? e : (8 + e));
        b.h[e] = (k < K) ? sW[col_lo * KP + k] : (__bf16)0.f;
      }
    }
    acc = __builtin_amdgcn_wmma_f32_16x16x32_bf16(
        false, a.v, false, b.v, (short)0, acc, false, false);
  }

  const int   col = colbase + col_lo;
  const float bv  = bias ? bias[col] : 0.f;
#pragma unroll
  for (int v = 0; v < 8; ++v) {
    int m = rowbase + v + half * 8;         // C/D row layout
    if (m < n) {
      float val = acc[v] + bv;
      if (RELU)  val = fmaxf(val, 0.f);
      if (ADDPE) val += pe[(long)m * 32 + (col & 31)];  // [pe,pe,pe] tiling
      Y[(long)m * M + col] = val;
    }
  }
}

// ---------------------------------------------------------------------------
__global__ void zero_f32v4(float4* __restrict__ p, long n4) {
  long i = (long)blockIdx.x * blockDim.x + threadIdx.x;
  if (i < n4) p[i] = make_float4(0.f, 0.f, 0.f, 0.f);
}

// agg[dst[e], f] += x[src[e], f]    (float4 gather + 4 atomic f32 adds)
template <int D>
__global__ void scatter_add(const float* __restrict__ x,
                            const int* __restrict__ src,
                            const int* __restrict__ dst,
                            float* __restrict__ agg, int E) {
  constexpr int D4 = D / 4;
  long t = (long)blockIdx.x * blockDim.x + threadIdx.x;
  if (t >= (long)E * D4) return;
  int e = (int)(t / D4), f4 = (int)(t - (long)e * D4);
  float4 v = *(const float4*)(x + (long)src[e] * D + 4 * f4);
  float* o = agg + (long)dst[e] * D + 4 * f4;
  atomicAdd(o + 0, v.x); atomicAdd(o + 1, v.y);
  atomicAdd(o + 2, v.z); atomicAdd(o + 3, v.w);
}

// agg[d, f] += dinv[s]*dinv[d] * h[s, f]
template <int D>
__global__ void gcn_scatter(const float* __restrict__ h,
                            const int* __restrict__ src,
                            const int* __restrict__ dst,
                            const float* __restrict__ dinv,
                            float* __restrict__ agg, int E) {
  constexpr int D4 = D / 4;
  long t = (long)blockIdx.x * blockDim.x + threadIdx.x;
  if (t >= (long)E * D4) return;
  int e = (int)(t / D4), f4 = (int)(t - (long)e * D4);
  int s = src[e], d = dst[e];
  float nm = dinv[s] * dinv[d];
  float4 v = *(const float4*)(h + (long)s * D + 4 * f4);
  float* o = agg + (long)d * D + 4 * f4;
  atomicAdd(o + 0, nm * v.x); atomicAdd(o + 1, nm * v.y);
  atomicAdd(o + 2, nm * v.z); atomicAdd(o + 3, nm * v.w);
}

__global__ void deg_count(const int* __restrict__ dst, float* __restrict__ deg, int E) {
  int e = blockIdx.x * blockDim.x + threadIdx.x;
  if (e < E) atomicAdd(&deg[dst[e]], 1.0f);
}

__global__ void deg_inv(const float* __restrict__ degc, float* __restrict__ dinv,
                        float* __restrict__ rinv, int n) {
  int i = blockIdx.x * blockDim.x + threadIdx.x;
  if (i >= n) return;
  float d = 1.0f + degc[i];
  dinv[i] = rsqrtf(d);
  rinv[i] = 1.0f / d;
}

// out = act(agg + h*rinv[row] + bias)
template <int D>
__global__ void gcn_finish(const float* __restrict__ agg, const float* __restrict__ h,
                           const float* __restrict__ rinv, const float* __restrict__ bias,
                           float* __restrict__ out, int n, int relu) {
  long t = (long)blockIdx.x * blockDim.x + threadIdx.x;
  if (t >= (long)n * D) return;
  int row = (int)(t / D), f = (int)(t - (long)row * D);
  float v = agg[t] + h[t] * rinv[row] + bias[f];
  out[t] = relu ? fmaxf(v, 0.f) : v;
}

// z[i, c] += h2[i, c % 16]   (cat([h2]*4))
__global__ void add_tile4(float* __restrict__ z, const float* __restrict__ h2, int n) {
  long t = (long)blockIdx.x * blockDim.x + threadIdx.x;
  if (t >= (long)n * 64) return;
  int row = (int)(t >> 6), f = (int)(t & 63);
  z[t] += h2[(long)row * 16 + (f & 15)];
}

__global__ void seg_max(const float* __restrict__ gate, const int* __restrict__ batch,
                        unsigned* __restrict__ m, int n) {
  long t = (long)blockIdx.x * blockDim.x + threadIdx.x;
  if (t >= (long)n * 64) return;
  int row = (int)(t >> 6), f = (int)(t & 63);
  atomicMax(&m[(long)batch[row] * 64 + f], f2ord(gate[t]));
}

__global__ void seg_exp(const float* __restrict__ gate, const int* __restrict__ batch,
                        const unsigned* __restrict__ m, float* __restrict__ ex,
                        float* __restrict__ ssum, int n) {
  long t = (long)blockIdx.x * blockDim.x + threadIdx.x;
  if (t >= (long)n * 64) return;
  int row = (int)(t >> 6), f = (int)(t & 63);
  int b = batch[row];
  float v = expf(gate[t] - ord2f(m[(long)b * 64 + f]));
  ex[t] = v;
  atomicAdd(&ssum[(long)b * 64 + f], v);
}

__global__ void seg_pool(const float* __restrict__ ex, const float* __restrict__ ssum,
                         const float* __restrict__ z, const int* __restrict__ batch,
                         float* __restrict__ pooled, int n) {
  long t = (long)blockIdx.x * blockDim.x + threadIdx.x;
  if (t >= (long)n * 64) return;
  int row = (int)(t >> 6), f = (int)(t & 63);
  int b = batch[row];
  float alpha = ex[t] / ssum[(long)b * 64 + f];
  atomicAdd(&pooled[(long)b * 64 + f], alpha * z[t]);
}

__global__ void seg_gather(const float* __restrict__ pooled, const int* __restrict__ batch,
                           float* __restrict__ zn, int n) {
  long t = (long)blockIdx.x * blockDim.x + threadIdx.x;
  if (t >= (long)n * 64) return;
  int row = (int)(t >> 6), f = (int)(t & 63);
  zn[t] = pooled[(long)batch[row] * 64 + f];
}

// ---------------------------------------------------------------------------
extern "C" void kernel_launch(void* const* d_in, const int* in_sizes, int n_in,
                              void* d_out, int out_size, void* d_ws, size_t ws_size,
                              hipStream_t stream) {
  const float* x1    = (const float*)d_in[0];
  const float* x2    = (const float*)d_in[1];
  const float* pe    = (const float*)d_in[2];
  const int*   ei    = (const int*)d_in[3];
  const int*   batch = (const int*)d_in[4];
  const int*   src   = ei;
  const int*   dst   = ei + N_EDGES;

  const float* embed_W = (const float*)d_in[5],  *embed_b = (const float*)d_in[6];
  const float* mpf_W   = (const float*)d_in[7],  *mpf_b   = (const float*)d_in[8];
  const float* gcf_W   = (const float*)d_in[9],  *gcf_b   = (const float*)d_in[10];
  const float* mpc_W   = (const float*)d_in[11], *mpc_b   = (const float*)d_in[12];
  const float* gcc_W   = (const float*)d_in[13], *gcc_b   = (const float*)d_in[14];
  const float* a1_W    = (const float*)d_in[15], *a1_b    = (const float*)d_in[16];
  const float* a2_W    = (const float*)d_in[17], *a2_b    = (const float*)d_in[18];
  const float* gcd_W   = (const float*)d_in[19], *gcd_b   = (const float*)d_in[20];
  const float* d1_W    = (const float*)d_in[21], *d1_b    = (const float*)d_in[22];
  const float* d2_W    = (const float*)d_in[23], *d2_b    = (const float*)d_in[24];
  const float* d3_W    = (const float*)d_in[25], *d3_b    = (const float*)d_in[26];

  float* ws = (float*)d_ws;
  float* P0   = ws;  ws += (long)N_NODES * 288;
  float* P1   = ws;  ws += (long)N_NODES * 288;
  float* P2   = ws;  ws += (long)N_NODES * 288;
  float* Z    = ws;  ws += (long)N_NODES * 64;
  float* DEGC = ws;  ws += N_NODES;
  float* DINV = ws;  ws += N_NODES;
  float* RINV = ws;  ws += N_NODES;
  float* SM   = ws;  ws += N_GRAPHS * 64;   // ordered-uint max
  float* SS   = ws;  ws += N_GRAPHS * 64;
  float* PL   = ws;  ws += N_GRAPHS * 64;
  float* OUT  = (float*)d_out;

  auto zero = [&](float* p, long cnt) {
    long n4 = cnt / 4;   // all call sites pass multiples of 4
    if (n4) zero_f32v4<<<(unsigned)((n4 + 255) / 256), 256, 0, stream>>>((float4*)p, n4);
  };
  const dim3 gemm_blk(128);
  const unsigned gx = (N_NODES + 63) / 64;
  const unsigned gN64 = (unsigned)(((long)N_NODES * 64 + 255) / 256);

#define GEMM(K, M, RELU, ADDPE, Xp, Wp, Bp, Yp, PEp)                            \
  gemm_bf16_wmma<K, M, RELU, ADDPE><<<dim3(gx, (M) / 16), gemm_blk, 0, stream>>>( \
      Xp, Wp, Bp, Yp, N_NODES, PEp)

#define SADD(D, Xp, Ap)                                                          \
  scatter_add<D><<<(unsigned)(((long)N_EDGES * ((D) / 4) + 255) / 256), 256, 0,  \
                   stream>>>(Xp, src, dst, Ap, N_EDGES)

#define GSCAT(D, Hp, Ap)                                                         \
  gcn_scatter<D><<<(unsigned)(((long)N_EDGES * ((D) / 4) + 255) / 256), 256, 0,  \
                   stream>>>(Hp, src, dst, DINV, Ap, N_EDGES)

#define GFIN(D, Ap, Hp, Bp, Op, RELU)                                            \
  gcn_finish<D><<<(unsigned)(((long)N_NODES * (D) + 255) / 256), 256, 0,         \
                  stream>>>(Ap, Hp, RINV, Bp, Op, N_NODES, RELU)

  // ---- degrees (shared by all GCN layers)
  zero(DEGC, N_NODES);
  deg_count<<<(N_EDGES + 255) / 256, 256, 0, stream>>>(dst, DEGC, N_EDGES);
  deg_inv<<<(N_NODES + 255) / 256, 256, 0, stream>>>(DEGC, DINV, RINV, N_NODES);

  // ---- encoder feature path
  GEMM(288, 96, true, true, x1, embed_W, embed_b, P0, pe);   // h1 = relu(x1@W+b)+pe3
  zero(P1, (long)N_NODES * 96);  SADD(96, P0, P1);
  GEMM(96, 128, true, false, P1, mpf_W, mpf_b, P2, nullptr);
  GEMM(128, 64, false, false, P2, gcf_W, nullptr, P0, nullptr);   // hW
  zero(P1, (long)N_NODES * 64);  GSCAT(64, P0, P1);
  GFIN(64, P1, P0, gcf_b, Z, 1);                              // h1c -> Z

  // ---- encoder class path
  zero(P0, (long)N_NODES * 32);  SADD(32, x2, P0);
  GEMM(32, 16, true, false, P0, mpc_W, mpc_b, P1, nullptr);
  GEMM(16, 16, false, false, P1, gcc_W, nullptr, P0, nullptr);    // hW
  zero(P2, (long)N_NODES * 16);  GSCAT(16, P0, P2);
  GFIN(16, P2, P0, gcc_b, P1, 1);                             // h2b
  add_tile4<<<gN64, 256, 0, stream>>>(Z, P1, N_NODES);        // z = h1c + cat4(h2b)

  // ---- global attention pooling
  GEMM(64, 256, true, false, Z, a1_W, a1_b, P0, nullptr);
  GEMM(256, 64, false, false, P0, a2_W, a2_b, P1, nullptr);   // gate
  zero(SM, (long)N_GRAPHS * 64);                              // ord 0 < ord(any real)
  seg_max<<<gN64, 256, 0, stream>>>(P1, batch, (unsigned*)SM, N_NODES);
  zero(SS, (long)N_GRAPHS * 64);
  seg_exp<<<gN64, 256, 0, stream>>>(P1, batch, (const unsigned*)SM, P2, SS, N_NODES);
  zero(PL, (long)N_GRAPHS * 64);
  seg_pool<<<gN64, 256, 0, stream>>>(P2, SS, Z, batch, PL, N_NODES);
  seg_gather<<<gN64, 256, 0, stream>>>(PL, batch, P0, N_NODES);   // zn

  // ---- decoder
  GEMM(64, 64, false, false, P0, gcd_W, nullptr, P1, nullptr);    // hW
  zero(P2, (long)N_NODES * 64);  GSCAT(64, P1, P2);
  GFIN(64, P2, P1, gcd_b, P0, 1);                             // d
  zero(P1, (long)N_NODES * 64);  SADD(64, P0, P1);
  GEMM(64, 96, true, true, P1, d1_W, d1_b, P2, pe);           // relu(...) + pe3
  zero(P0, (long)N_NODES * 96);  SADD(96, P2, P0);
  GEMM(96, 96, true, false, P0, d2_W, d2_b, P1, nullptr);
  zero(P2, (long)N_NODES * 96);  SADD(96, P1, P2);
  GEMM(96, 288, false, false, P2, d3_W, d3_b, OUT, nullptr);  // final, no relu
}
